// MyGNN_86045374808469
// MI455X (gfx1250) — matrix operand order repaired and checked by reading, not compiled, not run on hardware
//
#include <hip/hip_runtime.h>

#define N_NODES  10000
#define N_EDGES  320000
#define N_GRAPHS 64

typedef __attribute__((ext_vector_type(16))) _Float16 v16h;
typedef __attribute__((ext_vector_type(8)))  float    v8f;

__device__ __forceinline__ float sigmoidf(float v) {
    return 1.0f / (1.0f + __expf(-v));
}

// ---------------------------------------------------------------------------
// EdgeConv1: m = sigmoid([x_dst, x_src-x_dst] @ W1 + b1), max-scatter to dst.
// Outputs are positive => float-bits atomicMax on zeroed buffer == segment max
// with 0 fill (matches reference's isneginf->0 behavior).
// ---------------------------------------------------------------------------
__global__ void edgeconv1_kernel(const float* __restrict__ x,
                                 const int* __restrict__ src,
                                 const int* __restrict__ dst,
                                 const float* __restrict__ W1,
                                 const float* __restrict__ b1,
                                 unsigned* __restrict__ h1u) {
    __shared__ float sW[6 * 128];
    __shared__ float sb[128];
    for (int i = threadIdx.x; i < 6 * 128; i += blockDim.x) sW[i] = W1[i];
    for (int i = threadIdx.x; i < 128; i += blockDim.x) sb[i] = b1[i];
    __syncthreads();

    int e = blockIdx.x * blockDim.x + threadIdx.x;
    if (e >= N_EDGES) return;
    int s = src[e], d = dst[e];
    float xd0 = x[d * 3 + 0], xd1 = x[d * 3 + 1], xd2 = x[d * 3 + 2];
    float f3 = x[s * 3 + 0] - xd0;
    float f4 = x[s * 3 + 1] - xd1;
    float f5 = x[s * 3 + 2] - xd2;
    unsigned* out = h1u + (size_t)d * 128;
    for (int j = 0; j < 128; ++j) {
        float acc = sb[j]
                  + xd0 * sW[0 * 128 + j] + xd1 * sW[1 * 128 + j] + xd2 * sW[2 * 128 + j]
                  + f3  * sW[3 * 128 + j] + f4  * sW[4 * 128 + j] + f5  * sW[5 * 128 + j];
        float m = sigmoidf(acc);
        atomicMax(&out[j], __float_as_uint(m));
    }
}

// ---------------------------------------------------------------------------
// Convert W2 [K=256][N=256] f32 -> f16 transposed [N][K] so WMMA B fragments
// are contiguous 128-bit loads.
// ---------------------------------------------------------------------------
__global__ void conv_w2_kernel(const float* __restrict__ W2,
                               _Float16* __restrict__ W2hT) {
    int idx = blockIdx.x * blockDim.x + threadIdx.x;   // 65536 elems
    int n = idx >> 8;
    int k = idx & 255;
    W2hT[n * 256 + k] = (_Float16)W2[k * 256 + n];
}

// ---------------------------------------------------------------------------
// EdgeConv2 via V_WMMA_F32_16X16X32_F16.
// Block = 256 threads (8 waves) handles 64 edges (4 M tiles) x all 256 outputs.
// A tile (64x256 f16, 32 KB) built in LDS. Each wave owns two 16-col N tiles;
// per N tile it loads all 8 B fragments (K=256) into registers ONCE and reuses
// them across the 4 M sub-tiles -> 32 WMMAs per B-load set, A streamed from
// LDS via ds_load_b128. Bias-seeded accumulator, sigmoid + atomicMax scatter.
// ---------------------------------------------------------------------------
__global__ void __launch_bounds__(256)
edgeconv2_wmma_kernel(const float* __restrict__ h1,
                      const int* __restrict__ src,
                      const int* __restrict__ dst,
                      const _Float16* __restrict__ W2hT,
                      const float* __restrict__ b2,
                      unsigned* __restrict__ h2u) {
    __shared__ alignas(16) _Float16 As[64 * 256];  // 32 KB
    __shared__ int dsts[64];

    const int tid = threadIdx.x;
    const int e0  = blockIdx.x * 64;

    // ---- build A tile: 64 rows x 256 features; 16 threads per row chunk ----
    {
        int rsub  = tid >> 4;   // 0..15
        int lanec = tid & 15;   // 0..15
#pragma unroll
        for (int mt = 0; mt < 4; ++mt) {
            int m = mt * 16 + rsub;            // 0..63
            int s = src[e0 + m];
            int d = dst[e0 + m];
            if (lanec == 0) dsts[m] = d;
            const float* hd = h1 + (size_t)d * 128;
            const float* hs = h1 + (size_t)s * 128;
#pragma unroll
            for (int it = 0; it < 16; ++it) {
                int k = lanec + it * 16;        // 0..255
                float v;
                if (k < 128) v = hd[k];
                else         v = hs[k - 128] - hd[k - 128];
                As[m * 256 + k] = (_Float16)v;
            }
        }
    }
    __syncthreads();

    const int wave = tid >> 5;    // 0..7
    const int lane = tid & 31;
    const int col  = lane & 15;   // C column / fragment row-index component
    const int hsel = lane >> 4;   // half-wave select

#pragma unroll
    for (int t = 0; t < 2; ++t) {
        const int nt   = wave + t * 8;     // N tile 0..15
        const int ncol = nt * 16 + col;

        // ---- load all 8 B fragments for this N tile into registers ----
        // B 32x16 f16 layout: halfs 0..7 -> K = kb + 8*hsel + i, halfs 8..15 -> K = kb + 16 + 8*hsel + i
        union { uint4 u[2]; v16h h; } ub[8];
        const _Float16* brow = W2hT + (size_t)ncol * 256 + hsel * 8;
#pragma unroll
        for (int kk = 0; kk < 8; ++kk) {
            const uint4* bp = (const uint4*)(brow + kk * 32);
            ub[kk].u[0] = bp[0];
            ub[kk].u[1] = *(const uint4*)(brow + kk * 32 + 16);
        }
        float bias = b2[ncol];

#pragma unroll
        for (int mt = 0; mt < 4; ++mt) {
            v8f c;
#pragma unroll
            for (int r = 0; r < 8; ++r) c[r] = bias;

            const _Float16* arow = &As[(mt * 16 + col) * 256 + hsel * 8];
#pragma unroll
            for (int kk = 0; kk < 8; ++kk) {
                // A 16x32 f16 layout: halfs 0..7 -> K = kb + 8*hsel + i,
                //                     halfs 8..15 -> K = kb + 16 + 8*hsel + i
                union { uint4 u[2]; v16h h; } ua;
                ua.u[0] = *(const uint4*)(arow + kk * 32);
                ua.u[1] = *(const uint4*)(arow + kk * 32 + 16);
                c = __builtin_amdgcn_wmma_f32_16x16x32_f16(
                        false, ua.h, false, ub[kk].h, (short)0, c, false, false);
            }

            // C layout: VGPR r -> row r + 8*hsel (within M sub-tile), col = col
#pragma unroll
            for (int r = 0; r < 8; ++r) {
                int row = mt * 16 + r + 8 * hsel;
                float v = sigmoidf(c[r]);
                atomicMax(&h2u[(size_t)dsts[row] * 256 + nt * 16 + col],
                          __float_as_uint(v));
            }
        }
    }
}

// ---------------------------------------------------------------------------
// Graph max-pool: per node, atomicMax its 256 features into its graph row.
// ---------------------------------------------------------------------------
__global__ void pool_max_kernel(const float* __restrict__ h2,
                                const int* __restrict__ batch,
                                unsigned* __restrict__ gu) {
    int i = blockIdx.x;             // node
    int g = batch[i];
    const float* row = h2 + (size_t)i * 256;
    unsigned* grow = gu + (size_t)g * 256;
    for (int c = threadIdx.x; c < 256; c += blockDim.x)
        atomicMax(&grow[c], __float_as_uint(row[c]));
}

// ---------------------------------------------------------------------------
// Head: out = sigmoid(g @ W3 + b3) @ W4 + b4    (one block per graph)
// ---------------------------------------------------------------------------
__global__ void head_kernel(const float* __restrict__ g,
                            const float* __restrict__ W3,
                            const float* __restrict__ b3,
                            const float* __restrict__ W4,
                            const float* __restrict__ b4,
                            float* __restrict__ out) {
    __shared__ float t[128];
    int gi = blockIdx.x;
    int j  = threadIdx.x;           // blockDim = 128
    const float* grow = g + (size_t)gi * 256;
    float acc = b3[j];
    for (int k = 0; k < 256; ++k) acc += grow[k] * W3[k * 128 + j];
    t[j] = sigmoidf(acc);
    __syncthreads();
    if (j < 10) {
        float o = b4[j];
        for (int k = 0; k < 128; ++k) o += t[k] * W4[k * 10 + j];
        out[gi * 10 + j] = o;
    }
}

// ---------------------------------------------------------------------------
extern "C" void kernel_launch(void* const* d_in, const int* in_sizes, int n_in,
                              void* d_out, int out_size, void* d_ws, size_t ws_size,
                              hipStream_t stream) {
    (void)in_sizes; (void)n_in; (void)out_size; (void)ws_size;

    const float* x     = (const float*)d_in[0];
    const int*   ei    = (const int*)d_in[1];
    const int*   batch = (const int*)d_in[2];
    const float* W1    = (const float*)d_in[3];
    const float* b1    = (const float*)d_in[4];
    const float* W2    = (const float*)d_in[5];
    const float* b2    = (const float*)d_in[6];
    const float* W3    = (const float*)d_in[7];
    const float* b3    = (const float*)d_in[8];
    const float* W4    = (const float*)d_in[9];
    const float* b4    = (const float*)d_in[10];
    const int* src = ei;
    const int* dst = ei + N_EDGES;

    // workspace carve-up (256B aligned)
    char* ws = (char*)d_ws;
    size_t off = 0;
    auto carve = [&](size_t bytes) -> void* {
        void* p = ws + off;
        off = (off + bytes + 255) & ~(size_t)255;
        return p;
    };
    float*    h1   = (float*)carve((size_t)N_NODES * 128 * sizeof(float));
    float*    h2   = (float*)carve((size_t)N_NODES * 256 * sizeof(float));
    float*    g    = (float*)carve((size_t)N_GRAPHS * 256 * sizeof(float));
    _Float16* W2hT = (_Float16*)carve((size_t)256 * 256 * sizeof(_Float16));

    hipMemsetAsync(h1, 0, (size_t)N_NODES * 128 * sizeof(float), stream);
    hipMemsetAsync(h2, 0, (size_t)N_NODES * 256 * sizeof(float), stream);
    hipMemsetAsync(g,  0, (size_t)N_GRAPHS * 256 * sizeof(float), stream);

    conv_w2_kernel<<<256, 256, 0, stream>>>(W2, W2hT);
    edgeconv1_kernel<<<(N_EDGES + 255) / 256, 256, 0, stream>>>(
        x, src, dst, W1, b1, (unsigned*)h1);
    edgeconv2_wmma_kernel<<<N_EDGES / 64, 256, 0, stream>>>(
        h1, src, dst, W2hT, b2, (unsigned*)h2);
    pool_max_kernel<<<N_NODES, 256, 0, stream>>>(h2, batch, (unsigned*)g);
    head_kernel<<<N_GRAPHS, 128, 0, stream>>>(g, W3, b3, W4, b4, (float*)d_out);
}
